// GNN_24910810316997
// MI455X (gfx1250) — compile-verified, matrix operand order checked by reading, not run on hardware
//
#include <hip/hip_runtime.h>

#define N_NODES 9216
#define TOPK    3
#define C_IN    256
#define K_DIM   6400     // 256 * 5 * 5
#define HW_     480
#define C_OUT   256
#define G96     96
#define R_      8

typedef __attribute__((ext_vector_type(16))) __bf16 v16bf;
typedef __attribute__((ext_vector_type(8)))  float  v8f;

__device__ __forceinline__ unsigned short bf_bits(float f) {
  union { float f; unsigned u; } a; a.f = f;
  return (unsigned short)((a.u + 0x7fffu + ((a.u >> 16) & 1u)) >> 16);  // RNE
}
__device__ __forceinline__ __bf16 bits_bf(unsigned short s) {
  union { unsigned short s; __bf16 b; } o; o.s = s; return o.b;
}

// ---------------------------------------------------------------------------
// 1) Patch-embed GEMM: emb[N,256] = relu(patches[N,6400] @ conv_w^T + b)
//    8 waves / workgroup share one 16x32 A tile in LDS (8x less x traffic);
//    each wave owns a 16-channel B tile and one 16x16 C tile.
//    K stepped by 32; bf16 WMMA, f32 accumulate.
// ---------------------------------------------------------------------------
__global__ __launch_bounds__(256) void k_patch_embed(
    const float* __restrict__ x, const float* __restrict__ w,
    const float* __restrict__ b, float* __restrict__ emb) {
  const int tid  = threadIdx.x;
  const int lane = tid & 31;
  const int wv   = tid >> 5;                       // wave 0..7
  const int lo   = lane & 15, hi = lane >> 4;
  const int m0   = blockIdx.x << 4;                // 16 node rows per block
  const int n0   = (blockIdx.y << 7) + (wv << 4);  // 16 out-channels per wave

  __shared__ unsigned short lA[512];        // 16 rows x 32 k (bf16 bits), shared
  __shared__ unsigned short lB[8 * 512];    // per-wave 16 cols x 32 k
  unsigned short* lBw = &lB[wv * 512];

  // A staging: 512 elements / 256 threads -> 2 per thread (same k-col, rows r, r+8)
  const int er0 = tid >> 5, ekc = tid & 31;        // row 0..7, kcol 0..31
  const int er1 = er0 + 8;
  const int p0 = m0 + er0, p1 = m0 + er1;
  const int pr0 = p0 / G96, pc0 = p0 - pr0 * G96;
  const int pr1 = p1 / G96, pc1 = p1 - pr1 * G96;
  const float* xb0 = x + (size_t)(pr0 * 5) * HW_ + pc0 * 5;
  const float* xb1 = x + (size_t)(pr1 * 5) * HW_ + pc1 * 5;

  const float* wrow = w + (size_t)(n0 + lo) * K_DIM;

  v8f acc;
  const float bv = b[n0 + lo];
#pragma unroll
  for (int r = 0; r < 8; ++r) acc[r] = bv;

  for (int k = 0; k < K_DIM; k += 32) {
    // ---- stage A (whole block cooperates) ----
    {
      const int kk = k + ekc;
      const int c = kk / 25, rem = kk - c * 25;
      const int i = rem / 5, j = rem - i * 5;
      const size_t off = (size_t)c * (HW_ * HW_) + i * HW_ + j;
      lA[er0 * 32 + ekc] = bf_bits(xb0[off]);
      lA[er1 * 32 + ekc] = bf_bits(xb1[off]);
    }
    // ---- stage this wave's B tile (16 contiguous k per lane) ----
#pragma unroll
    for (int t = 0; t < 16; ++t)
      lBw[lo * 32 + hi * 16 + t] = bf_bits(wrow[k + hi * 16 + t]);
    __syncthreads();

    // fragment assembly per documented 16-bit A layout:
    // lanes 0-15: K = {0..7, 16..23}; lanes 16-31: +8
    v16bf af, bfm;
#pragma unroll
    for (int e = 0; e < 16; ++e) {
      const int koff = ((e < 8) ? e : e + 8) + hi * 8;
      af[e]  = bits_bf(lA[lo * 32 + koff]);
      bfm[e] = bits_bf(lBw[lo * 32 + koff]);
    }
    acc = __builtin_amdgcn_wmma_f32_16x16x32_bf16(
        false, af, false, bfm, (short)0, acc, false, false);
    __syncthreads();

    if (k + 32 < K_DIM) __builtin_prefetch(wrow + k + 32, 0, 1);
  }

  // C layout: VGPR r -> M = r + 8*hi, N = lo
#pragma unroll
  for (int r = 0; r < 8; ++r) {
    float v = acc[r];
    v = v > 0.f ? v : 0.f;   // relu
    emb[(size_t)(m0 + r + hi * 8) * C_OUT + (n0 + lo)] = v;
  }
}

// ---------------------------------------------------------------------------
// 2) fc2: emb_r[N,8] = relu(emb[:, :8] @ fc2_w + b)
// ---------------------------------------------------------------------------
__global__ void k_fc2(const float* __restrict__ emb, const float* __restrict__ w,
                      const float* __restrict__ b, float* __restrict__ out) {
  const int t = blockIdx.x * blockDim.x + threadIdx.x;
  if (t >= N_NODES * R_) return;
  const int n = t / R_, o = t - n * R_;
  float s = b[o];
  for (int i = 0; i < R_; ++i) s += emb[(size_t)n * C_OUT + i] * w[i * R_ + o];
  out[t] = s > 0.f ? s : 0.f;
}

// ---------------------------------------------------------------------------
// 3) similarity top-k (K=3) on emb_r (inner dim 8)
// ---------------------------------------------------------------------------
__global__ void k_topk_sim(const float* __restrict__ er,
                           int* __restrict__ idx, float* __restrict__ val) {
  const int n = blockIdx.x * blockDim.x + threadIdx.x;
  if (n >= N_NODES) return;
  float e[R_];
#pragma unroll
  for (int d = 0; d < R_; ++d) e[d] = er[(size_t)n * R_ + d];
  float v0 = -3e38f, v1 = -3e38f, v2 = -3e38f;
  int   i0 = 0, i1 = 0, i2 = 0;
  for (int j = 0; j < N_NODES; ++j) {
    float s = 0.f;
#pragma unroll
    for (int d = 0; d < R_; ++d) s += e[d] * er[(size_t)j * R_ + d];
    if (s > v0)      { v2=v1; i2=i1; v1=v0; i1=i0; v0=s; i0=j; }
    else if (s > v1) { v2=v1; i2=i1; v1=s;  i1=j; }
    else if (s > v2) { v2=s;  i2=j; }
  }
  idx[n*TOPK+0]=i0; idx[n*TOPK+1]=i1; idx[n*TOPK+2]=i2;
  val[n*TOPK+0]=v0; val[n*TOPK+1]=v1; val[n*TOPK+2]=v2;
}

// ---------------------------------------------------------------------------
// 4) spatial top-k: nearest patch centers; store distance as "val"
// ---------------------------------------------------------------------------
__global__ void k_topk_spatial(int* __restrict__ idx, float* __restrict__ val) {
  const int n = blockIdx.x * blockDim.x + threadIdx.x;
  if (n >= N_NODES) return;
  const int pr = n / G96, pc = n - pr * G96;
  float v0 = -3e38f, v1 = -3e38f, v2 = -3e38f;
  int   i0 = 0, i1 = 0, i2 = 0;
  for (int jr = 0; jr < G96; ++jr)
    for (int jc = 0; jc < G96; ++jc) {
      const float dy = (float)((pr - jr) * 5);
      const float dx = (float)((pc - jc) * 5);
      const float s  = -sqrtf(dy * dy + dx * dx);   // top_k of -dist
      const int   j  = jr * G96 + jc;
      if (s > v0)      { v2=v1; i2=i1; v1=v0; i1=i0; v0=s; i0=j; }
      else if (s > v1) { v2=v1; i2=i1; v1=s;  i1=j; }
      else if (s > v2) { v2=s;  i2=j; }
    }
  idx[n*TOPK+0]=i0; idx[n*TOPK+1]=i1; idx[n*TOPK+2]=i2;
  val[n*TOPK+0]=-v0; val[n*TOPK+1]=-v1; val[n*TOPK+2]=-v2;  // vals = +dist
}

// ---------------------------------------------------------------------------
// 5) GATv2: projections xl = x@wl+bl, xr = x@wr+br
// ---------------------------------------------------------------------------
__global__ void k_gat_proj(const float* __restrict__ xin,
                           const float* __restrict__ wl, const float* __restrict__ bl,
                           const float* __restrict__ wr, const float* __restrict__ br,
                           float* __restrict__ xl, float* __restrict__ xr,
                           int Fin, int Hout) {
  const int t = blockIdx.x * blockDim.x + threadIdx.x;
  if (t >= N_NODES * Hout) return;
  const int n = t / Hout, o = t - n * Hout;
  float sl = bl[o], sr = br[o];
  for (int i = 0; i < Fin; ++i) {
    const float xv = xin[(size_t)n * Fin + i];
    sl += xv * wl[i * Hout + o];
    sr += xv * wr[i * Hout + o];
  }
  xl[t] = sl; xr[t] = sr;
}

// ---------------------------------------------------------------------------
// 6) GATv2 per-node aggregate. Incoming edges of node n are exactly its own
//    K top-k candidates (masked: idx>=n && val!=0 && idx!=n) + self loop.
//    One thread per (node, head); local softmax; elu(out + bias).
// ---------------------------------------------------------------------------
__global__ void k_gat_node(const int* __restrict__ idx, const float* __restrict__ val,
                           const float* __restrict__ xl, const float* __restrict__ xr,
                           const float* __restrict__ att, const float* __restrict__ bias,
                           float* __restrict__ out, int H) {
  const int t = blockIdx.x * blockDim.x + threadIdx.x;
  if (t >= N_NODES * H) return;
  const int n = t / H, h = t - n * H;
  const int Hout = H * R_;

  float xrv[R_], av[R_];
#pragma unroll
  for (int d = 0; d < R_; ++d) {
    xrv[d] = xr[(size_t)n * Hout + h * R_ + d];
    av[d]  = att[h * R_ + d];
  }

  int   es[TOPK + 1];
  float sc[TOPK + 1];
  float mx = -3e38f;
  for (int j = 0; j <= TOPK; ++j) {
    int s; bool keep;
    if (j < TOPK) {
      const int e = n * TOPK + j;
      s = idx[e];
      keep = (s >= n) && (val[e] != 0.f) && (s != n);
    } else { s = n; keep = true; }
    if (!keep) { es[j] = -1; continue; }
    float sco = 0.f;
#pragma unroll
    for (int d = 0; d < R_; ++d) {
      float v = xl[(size_t)s * Hout + h * R_ + d] + xrv[d];
      v = v > 0.f ? v : 0.2f * v;     // leaky_relu(0.2)
      sco += av[d] * v;
    }
    es[j] = s; sc[j] = sco;
    if (sco > mx) mx = sco;
  }

  float denom = 0.f, num[R_];
#pragma unroll
  for (int d = 0; d < R_; ++d) num[d] = 0.f;
  for (int j = 0; j <= TOPK; ++j) {
    if (es[j] < 0) continue;
    const float pj = __expf(sc[j] - mx);
    denom += pj;
#pragma unroll
    for (int d = 0; d < R_; ++d)
      num[d] += xl[(size_t)es[j] * Hout + h * R_ + d] * pj;
  }
#pragma unroll
  for (int d = 0; d < R_; ++d) {
    const float v = num[d] / denom + bias[h * R_ + d];
    out[(size_t)n * Hout + h * R_ + d] = v > 0.f ? v : (__expf(v) - 1.f);  // elu
  }
}

// ---------------------------------------------------------------------------
// 7) fc3 + residual: out = relu(concat(h2,h4) @ fc3_w + b) + h4
// ---------------------------------------------------------------------------
__global__ void k_fc3(const float* __restrict__ h2, const float* __restrict__ h4,
                      const float* __restrict__ w, const float* __restrict__ b,
                      float* __restrict__ out) {
  const int t = blockIdx.x * blockDim.x + threadIdx.x;
  if (t >= N_NODES * R_) return;
  const int n = t / R_, o = t - n * R_;
  float s = b[o];
  for (int i = 0; i < R_; ++i) s += h2[(size_t)n * R_ + i] * w[i * R_ + o];
  for (int i = 0; i < R_; ++i) s += h4[(size_t)n * R_ + i] * w[(R_ + i) * R_ + o];
  const float r = s > 0.f ? s : 0.f;
  out[t] = r + h4[(size_t)n * R_ + o];
}

// ---------------------------------------------------------------------------
static void run_gat(hipStream_t st, const float* xin, int Fin, int H,
                    const float* const* prm,   // wl, bl, wr, br, att, bias
                    const int* idx, const float* val,
                    float* xl, float* xr, float* out) {
  const int Hout = H * R_;
  const int T1 = N_NODES * Hout;
  k_gat_proj<<<(T1 + 255) / 256, 256, 0, st>>>(xin, prm[0], prm[1], prm[2], prm[3],
                                               xl, xr, Fin, Hout);
  const int T2 = N_NODES * H;
  k_gat_node<<<(T2 + 255) / 256, 256, 0, st>>>(idx, val, xl, xr, prm[4], prm[5], out, H);
}

extern "C" void kernel_launch(void* const* d_in, const int* in_sizes, int n_in,
                              void* d_out, int out_size, void* d_ws, size_t ws_size,
                              hipStream_t stream) {
  (void)in_sizes; (void)n_in; (void)out_size; (void)ws_size;
  const float* x      = (const float*)d_in[0];
  const float* conv_w = (const float*)d_in[1];
  const float* conv_b = (const float*)d_in[2];
  const float* fc2_w  = (const float*)d_in[3];
  const float* fc2_b  = (const float*)d_in[4];
  const float* fc3_w  = (const float*)d_in[5];
  const float* fc3_b  = (const float*)d_in[6];
  const float* g[4][6];
  for (int l = 0; l < 4; ++l)
    for (int q = 0; q < 6; ++q) g[l][q] = (const float*)d_in[7 + l * 6 + q];

  // workspace carve-out (256B aligned)
  char* wp = (char*)d_ws;
  auto alloc = [&](size_t bytes) -> void* {
    void* r = (void*)wp; wp += (bytes + 255) & ~(size_t)255; return r;
  };
  float* emb  = (float*)alloc(sizeof(float) * (size_t)N_NODES * C_OUT);
  float* embr = (float*)alloc(sizeof(float) * N_NODES * R_);
  int*   sidx = (int*)  alloc(sizeof(int)   * N_NODES * TOPK);
  float* sval = (float*)alloc(sizeof(float) * N_NODES * TOPK);
  int*   pidx = (int*)  alloc(sizeof(int)   * N_NODES * TOPK);
  float* pval = (float*)alloc(sizeof(float) * N_NODES * TOPK);
  float* xl   = (float*)alloc(sizeof(float) * (size_t)N_NODES * 64);
  float* xr   = (float*)alloc(sizeof(float) * (size_t)N_NODES * 64);
  float* h1   = (float*)alloc(sizeof(float) * (size_t)N_NODES * 64);
  float* h2   = (float*)alloc(sizeof(float) * N_NODES * R_);
  float* h3   = (float*)alloc(sizeof(float) * (size_t)N_NODES * 64);
  float* h4   = (float*)alloc(sizeof(float) * N_NODES * R_);

  // 1) WMMA patch-embed GEMM (dominant compute): 8 waves/block share A tile
  dim3 gE(N_NODES / 16, C_OUT / 128);
  k_patch_embed<<<gE, 256, 0, stream>>>(x, conv_w, conv_b, emb);

  // 2) fc2
  const int T = N_NODES * R_;
  k_fc2<<<(T + 255) / 256, 256, 0, stream>>>(emb, fc2_w, fc2_b, embr);

  // 3) graphs
  k_topk_sim<<<(N_NODES + 255) / 256, 256, 0, stream>>>(embr, sidx, sval);
  k_topk_spatial<<<(N_NODES + 255) / 256, 256, 0, stream>>>(pidx, pval);

  // 4) GAT stack
  run_gat(stream, embr, 8,  8, g[0], sidx, sval, xl, xr, h1);  // g1
  run_gat(stream, h1,   64, 1, g[1], sidx, sval, xl, xr, h2);  // g2
  run_gat(stream, embr, 8,  8, g[2], pidx, pval, xl, xr, h3);  // g3
  run_gat(stream, h3,   64, 1, g[3], pidx, pval, xl, xr, h4);  // g4

  // 5) fuse + residual -> d_out
  k_fc3<<<(T + 255) / 256, 256, 0, stream>>>(h2, h4, fc3_w, fc3_b, (float*)d_out);
}
